// EventPillarFeatureNet_386547057186
// MI455X (gfx1250) — matrix operand order repaired and checked by reading, not compiled, not run on hardware
//
#include <hip/hip_runtime.h>
#include <hip/hip_bf16.h>

// ---------------- problem constants (from reference) ----------------
#define GXC 116
#define GYC 87
#define BC  4
#define NSEGC (BC * GXC * GYC)   // 40368
#define EPSC 1e-3f
#define ENC_NEGINF 0x007FFFFFu   // order-encoding of -inf

typedef __attribute__((ext_vector_type(16))) __bf16 v16bf;
typedef __attribute__((ext_vector_type(8)))  float  v8f;

// order-preserving float<->uint encoding so segment_max == atomicMax(u32)
__device__ __forceinline__ unsigned encf(float f) {
    unsigned u = __float_as_uint(f);
    return (u & 0x80000000u) ? ~u : (u | 0x80000000u);
}
__device__ __forceinline__ float decf(unsigned u) {
    unsigned v = (u & 0x80000000u) ? (u & 0x7FFFFFFFu) : ~u;
    return __uint_as_float(v);
}
__device__ __forceinline__ float swishf(float h) {
    return h / (1.0f + __expf(-h));
}

// ------------------------------------------------------------------
// K0: initialize workspace (runs every launch -> deterministic)
// ------------------------------------------------------------------
__global__ void k_init(unsigned* cnt, unsigned* m0e, unsigned* m1e,
                       float* s0, float* q0, float* s1, float* q1) {
    int i = blockIdx.x * blockDim.x + threadIdx.x;
    int stride = gridDim.x * blockDim.x;
    for (int j = i; j < NSEGC * 128; j += stride) m1e[j] = ENC_NEGINF;
    for (int j = i; j < NSEGC * 32;  j += stride) m0e[j] = ENC_NEGINF;
    for (int j = i; j < NSEGC;       j += stride) cnt[j] = 0u;
    if (i < 32)  { s0[i] = 0.f; q0[i] = 0.f; }
    if (i < 128) { s1[i] = 0.f; q1[i] = 0.f; }
}

// ------------------------------------------------------------------
// K1: voxelize + h0 = feats@W0 raw; accumulate BN0 sum/sumsq
// ------------------------------------------------------------------
__global__ void k_stage1(const float* __restrict__ pts,
                         const float* __restrict__ W0, int N,
                         int* __restrict__ inv, unsigned* __restrict__ cnt,
                         float* __restrict__ s0, float* __restrict__ q0) {
    __shared__ float ls[32], lq[32];
    if (threadIdx.x < 32) { ls[threadIdx.x] = 0.f; lq[threadIdx.x] = 0.f; }
    __syncthreads();
    int i = blockIdx.x * blockDim.x + threadIdx.x;
    if (i < N) {
        float b = pts[i*5+0], x = pts[i*5+1], y = pts[i*5+2];
        float t = pts[i*5+3], pol = pts[i*5+4];
        int cx = (int)floorf(x * (1.0f/3.0f));
        int cy = (int)floorf(y * (1.0f/3.0f));
        int seg = (((int)b) * GXC + cx) * GYC + cy;
        inv[i] = seg;
        atomicOr(&cnt[seg], 1u);
        float f0 = x * (1.0f/346.0f), f1 = y * (1.0f/260.0f);
        float f2 = t * (1.0f/200.0f), f3 = pol;
        #pragma unroll
        for (int j = 0; j < 32; ++j) {
            float h = f0*W0[j] + f1*W0[32+j] + f2*W0[64+j] + f3*W0[96+j];
            atomicAdd(&ls[j], h);
            atomicAdd(&lq[j], h*h);
        }
    }
    __syncthreads();
    if (threadIdx.x < 32) {
        atomicAdd(&s0[threadIdx.x], ls[threadIdx.x]);
        atomicAdd(&q0[threadIdx.x], lq[threadIdx.x]);
    }
}

// ------------------------------------------------------------------
// K2: fold BN stats into per-channel scale/shift
// ------------------------------------------------------------------
__global__ void k_bnfin(const float* __restrict__ s, const float* __restrict__ q,
                        const float* __restrict__ gamma, const float* __restrict__ beta,
                        int C, float invN, float* __restrict__ scale,
                        float* __restrict__ shift) {
    int j = blockIdx.x * blockDim.x + threadIdx.x;
    if (j < C) {
        float mu  = s[j] * invN;
        float var = q[j] * invN - mu * mu;
        float sc  = rsqrtf(var + EPSC) * gamma[j];
        scale[j] = sc;
        shift[j] = beta[j] - mu * sc;
    }
}

// ------------------------------------------------------------------
// K3: recompute h0, apply BN0+swish, store bf16 activations,
//     segment-max into m0 (encoded u32 atomics; m0 is L2-resident)
// ------------------------------------------------------------------
__global__ void k_stage2(const float* __restrict__ pts,
                         const float* __restrict__ W0,
                         const float* __restrict__ sc, const float* __restrict__ sh,
                         int N, const int* __restrict__ inv,
                         __bf16* __restrict__ h0act, unsigned* __restrict__ m0e) {
    int i = blockIdx.x * blockDim.x + threadIdx.x;
    if (i >= N) return;
    float x = pts[i*5+1], y = pts[i*5+2], t = pts[i*5+3], pol = pts[i*5+4];
    float f0 = x * (1.0f/346.0f), f1 = y * (1.0f/260.0f);
    float f2 = t * (1.0f/200.0f), f3 = pol;
    int seg = inv[i];
    #pragma unroll
    for (int j = 0; j < 32; ++j) {
        float h = f0*W0[j] + f1*W0[32+j] + f2*W0[64+j] + f3*W0[96+j];
        h = h * sc[j] + sh[j];
        float a = swishf(h);
        h0act[(size_t)i*32 + j] = (__bf16)a;
        atomicMax(&m0e[seg*32 + j], encf(a));
    }
}

// ------------------------------------------------------------------
// WMMA fragment builders — lane layouts per CDNA5 ISA §7.12.2
// A (16x32 bf16): row M = lane&15; lane<16 owns K{0..7,16..23},
//                 lane>=16 owns K{8..15,24..31}; 2 elems / VGPR.
//                 Each 8-K run is contiguous in LDS -> 2x ds_load_b128.
// B (32x16 bf16): col N = lane&15; lane<16 owns K{0..7,16..23},
//                 lane>=16 owns K{8..15,24..31} (same order as A).
//                 Tile-invariant -> built once per block from global W1.
// ------------------------------------------------------------------
__device__ __forceinline__ v16bf load_A_frag(const __bf16* lx /*16x64*/,
                                             int kbase, int lane) {
    const int row = lane & 15;
    const int kh  = (lane >> 4) * 8;
    union { uint4 u; __bf16 h[8]; } lo, hi;
    lo.u = *(const uint4*)&lx[row*64 + kbase + kh];        // K = kbase+kh+0..7
    hi.u = *(const uint4*)&lx[row*64 + kbase + 16 + kh];   // K = kbase+16+kh+0..7
    v16bf a;
    #pragma unroll
    for (int e = 0; e < 8; ++e) { a[e] = lo.h[e]; a[8+e] = hi.h[e]; }
    return a;
}
__device__ __forceinline__ v16bf make_B_frag(const float* __restrict__ W1,
                                             int kbase, int colbase, int lane) {
    const int col = colbase + (lane & 15);
    const int kh  = (lane >> 4) * 8;
    v16bf b;
    #pragma unroll
    for (int v = 0; v < 8; ++v) {
        int k = kbase + kh + (v < 4 ? 2*v : 8 + 2*v);
        b[2*v]   = (__bf16)W1[k*128 + col];
        b[2*v+1] = (__bf16)W1[(k+1)*128 + col];
    }
    return b;
}

// ------------------------------------------------------------------
// K4/K6: h1 = [h0 | m0[inv]] @ W1 via v_wmma_f32_16x16x32_bf16.
// PASS_B=false: accumulate BN1 sum/sumsq in registers (h1 never stored).
// PASS_B=true : recompute h1, BN1+swish epilogue, segment-max to m1.
// 128 threads = 4 waves; each wave owns 2 of 8 column tiles; W1
// fragments live in registers for the whole block (tile-invariant).
// ------------------------------------------------------------------
template <bool PASS_B>
__global__ void k_gemm(const __bf16* __restrict__ h0act,
                       const int* __restrict__ inv,
                       const unsigned* __restrict__ m0e,
                       const float* __restrict__ W1, int ntiles,
                       float* __restrict__ s1, float* __restrict__ q1,
                       const float* __restrict__ sc1, const float* __restrict__ sh1,
                       unsigned* __restrict__ m1e) {
    __shared__ __bf16 lx[16 * 64];    // x1 tile: 16 points x 64 features
    __shared__ int    linv[16];

    const int tid  = threadIdx.x;
    const int lane = tid & 31;
    const int wv   = tid >> 5;
    const int row  = tid >> 3;        // staging: 0..15
    const int cg   = tid & 7;         // staging: 8-col group 0..7

    // tile-invariant B fragments (W1 is 64x128; 32 VGPRs per lane)
    const int cb0 = (wv * 2) * 16, cb1 = (wv * 2 + 1) * 16;
    const v16bf b00 = make_B_frag(W1, 0,  cb0, lane);
    const v16bf b01 = make_B_frag(W1, 32, cb0, lane);
    const v16bf b10 = make_B_frag(W1, 0,  cb1, lane);
    const v16bf b11 = make_B_frag(W1, 32, cb1, lane);

    const int ch0 = cb0 + (lane & 15);
    const int ch1 = cb1 + (lane & 15);
    float sA0 = 0.f, qA0 = 0.f, sA1 = 0.f, qA1 = 0.f;  // pass-A stats regs
    float scl0 = 0.f, shf0 = 0.f, scl1 = 0.f, shf1 = 0.f;
    if (PASS_B) { scl0 = sc1[ch0]; shf0 = sh1[ch0];
                  scl1 = sc1[ch1]; shf1 = sh1[ch1]; }

    for (int tile = blockIdx.x; tile < ntiles; tile += gridDim.x) {
        if (tid < 16) linv[tid] = inv[tile * 16 + tid];
        __syncthreads();
        // stage x1 tile: one 16B packet per thread.
        if (cg < 4) {       // cols 0..31: straight bf16 copy, b128 in/out
            *(uint4*)&lx[row*64 + cg*8] =
                *(const uint4*)(h0act + (size_t)(tile*16 + row)*32 + cg*8);
        } else {            // cols 32..63: decode m0 gather (L2-resident)
            const int base = linv[row]*32 + (cg - 4)*8;
            union { uint4 u; __bf16 h[8]; } pkt;
            #pragma unroll
            for (int j = 0; j < 8; ++j) pkt.h[j] = (__bf16)decf(m0e[base + j]);
            *(uint4*)&lx[row*64 + cg*8] = pkt.u;
        }
        // prefetch next tile's h0act rows (global_prefetch_b8)
        const int nxt = tile + gridDim.x;
        if (nxt < ntiles)
            __builtin_prefetch(h0act + (size_t)(nxt*16 + row)*32 + cg*8, 0, 1);
        __syncthreads();

        const v16bf a0 = load_A_frag(lx, 0,  lane);
        const v16bf a1 = load_A_frag(lx, 32, lane);

        v8f acc0 = {}, acc1 = {};
        acc0 = __builtin_amdgcn_wmma_f32_16x16x32_bf16(
                   false, a0, false, b00, (short)0, acc0, false, false);
        acc0 = __builtin_amdgcn_wmma_f32_16x16x32_bf16(
                   false, a1, false, b01, (short)0, acc0, false, false);
        acc1 = __builtin_amdgcn_wmma_f32_16x16x32_bf16(
                   false, a0, false, b10, (short)0, acc1, false, false);
        acc1 = __builtin_amdgcn_wmma_f32_16x16x32_bf16(
                   false, a1, false, b11, (short)0, acc1, false, false);

        if (!PASS_B) {
            #pragma unroll
            for (int r = 0; r < 8; ++r) {
                float v0 = acc0[r], v1 = acc1[r];
                sA0 += v0; qA0 += v0*v0;
                sA1 += v1; qA1 += v1*v1;
            }
        } else {
            const int mofs = (lane >> 4) << 3;  // lanes 16-31 hold M=r+8
            #pragma unroll
            for (int r = 0; r < 8; ++r) {
                const int seg = linv[r + mofs];
                float h0v = acc0[r] * scl0 + shf0;
                float h1v = acc1[r] * scl1 + shf1;
                atomicMax(&m1e[seg * 128 + ch0], encf(swishf(h0v)));
                atomicMax(&m1e[seg * 128 + ch1], encf(swishf(h1v)));
            }
        }
        __syncthreads();
    }
    if (!PASS_B) {   // lanes L and L+16 share a channel; global atomics absorb it
        atomicAdd(&s1[ch0], sA0); atomicAdd(&q1[ch0], qA0);
        atomicAdd(&s1[ch1], sA1); atomicAdd(&q1[ch1], qA1);
    }
}

// ------------------------------------------------------------------
// K7: occupancy-masked scatter + align-corners bilinear 87x116 -> 224x224
// output layout (B=4, C=128, 224, 224); m1/cnt gathers are L2-resident
// ------------------------------------------------------------------
__global__ void k_resize(const unsigned* __restrict__ m1e,
                         const unsigned* __restrict__ cnt,
                         float* __restrict__ out, int total) {
    int idx = blockIdx.x * blockDim.x + threadIdx.x;
    if (idx >= total) return;
    int ox = idx % 224;
    int tmp = idx / 224;
    int oy = tmp % 224; tmp /= 224;
    int c  = tmp % 128;
    int b  = tmp / 128;

    float sy = oy * (86.0f  / 223.0f);
    float sx = ox * (115.0f / 223.0f);
    int y0 = (int)sy; int y1 = min(y0 + 1, 86);  float wy = sy - (float)y0;
    int x0 = (int)sx; int x1 = min(x0 + 1, 115); float wx = sx - (float)x0;

    auto val = [&](int gy, int gx) -> float {
        int seg = (b * GXC + gx) * GYC + gy;
        return cnt[seg] ? decf(m1e[seg * 128 + c]) : 0.0f;
    };
    float v00 = val(y0, x0), v01 = val(y0, x1);
    float v10 = val(y1, x0), v11 = val(y1, x1);
    float top = v00 * (1.0f - wx) + v01 * wx;
    float bot = v10 * (1.0f - wx) + v11 * wx;
    out[idx] = top * (1.0f - wy) + bot * wy;
}

// ------------------------------------------------------------------
extern "C" void kernel_launch(void* const* d_in, const int* in_sizes, int n_in,
                              void* d_out, int out_size, void* d_ws, size_t ws_size,
                              hipStream_t stream) {
    const float* points = (const float*)d_in[1];   // d_in[0] = fus (unused in reference)
    const float* W0 = (const float*)d_in[2];
    const float* g0 = (const float*)d_in[3];
    const float* b0 = (const float*)d_in[4];
    const float* W1 = (const float*)d_in[5];
    const float* g1 = (const float*)d_in[6];
    const float* b1 = (const float*)d_in[7];
    const int N = in_sizes[1] / 5;

    // workspace carve-out (256B aligned): ~95 MB total
    char* w = (char*)d_ws;
    auto carve = [&](size_t bytes) -> char* {
        char* p = w; w += (bytes + 255) & ~(size_t)255; return p;
    };
    int*      inv   = (int*)     carve((size_t)N * 4);
    unsigned* cnt   = (unsigned*)carve((size_t)NSEGC * 4);
    unsigned* m0e   = (unsigned*)carve((size_t)NSEGC * 32 * 4);
    unsigned* m1e   = (unsigned*)carve((size_t)NSEGC * 128 * 4);
    __bf16*   h0act = (__bf16*)  carve((size_t)N * 32 * 2);
    float* s0  = (float*)carve(32 * 4);
    float* q0  = (float*)carve(32 * 4);
    float* s1  = (float*)carve(128 * 4);
    float* q1  = (float*)carve(128 * 4);
    float* sc0 = (float*)carve(32 * 4);
    float* sh0 = (float*)carve(32 * 4);
    float* sc1 = (float*)carve(128 * 4);
    float* sh1 = (float*)carve(128 * 4);

    const float invN = 1.0f / (float)N;
    const int ntiles = N / 16;   // N = 1,000,000 -> 62,500 full tiles

    k_init<<<512, 256, 0, stream>>>(cnt, m0e, m1e, s0, q0, s1, q1);
    k_stage1<<<(N + 255) / 256, 256, 0, stream>>>(points, W0, N, inv, cnt, s0, q0);
    k_bnfin<<<1, 64, 0, stream>>>(s0, q0, g0, b0, 32, invN, sc0, sh0);
    k_stage2<<<(N + 255) / 256, 256, 0, stream>>>(points, W0, sc0, sh0, N, inv,
                                                  h0act, m0e);
    k_gemm<false><<<1024, 128, 0, stream>>>(h0act, inv, m0e, W1, ntiles,
                                            s1, q1, sc1, sh1, m1e);
    k_bnfin<<<1, 128, 0, stream>>>(s1, q1, g1, b1, 128, invN, sc1, sh1);
    k_gemm<true><<<1024, 128, 0, stream>>>(h0act, inv, m0e, W1, ntiles,
                                           s1, q1, sc1, sh1, m1e);
    const int total = BC * 128 * 224 * 224;
    k_resize<<<(total + 255) / 256, 256, 0, stream>>>(m1e, cnt, (float*)d_out, total);
}
